// LocalizationModel_49632642072694
// MI455X (gfx1250) — matrix operand order repaired and checked by reading, not compiled, not run on hardware
//
#include <hip/hip_runtime.h>

// ============================================================================
// MI455X (gfx1250) localization model.
//  - convs = implicit GEMM via v_wmma_f32_16x16x32_f16, compile-time shapes
//  - double-buffered LDS staging via async global->LDS b128 loads (ASYNCcnt)
//  - weights stored in natural K order within 32-chunks so both A and B
//    fragments are single contiguous 32B LDS reads (no v_mov shuffling);
//    permuting K identically on A and B leaves the WMMA dot product invariant
//  - instance-norm stats fused into conv epilogue via f32 atomics
// Workspace layout (bytes), total ~75.1 MiB (assumes ws_size >= 80 MiB):
//   [ 0,16M)  ACT0 f16 NHWC      [16M,32M) ACT1 f16 NHWC
//   [32M,64M) RAW  f32 NHWC      [64M,65M) WGT  f16 packed
//   [65M,66M) STATS f32          [66M,70M) PROJ_L f32 NHWC
//   [70M,74M) PROJ_M f32 NHWC    [74M,75M) COST f32
//   [75M,..)  ZPAD (256B zeros for OOB async loads)
// ============================================================================

typedef __attribute__((ext_vector_type(16))) _Float16 v16h;
typedef __attribute__((ext_vector_type(8)))  float    v8f;

template <int N> __device__ __forceinline__ void wait_async() {
  if constexpr (N == 0)      asm volatile("s_wait_asynccnt 0x0" ::: "memory");
  else if constexpr (N == 2) asm volatile("s_wait_asynccnt 0x2" ::: "memory");
  else                       asm volatile("s_wait_asynccnt 0x3" ::: "memory");
}

__device__ __forceinline__ void async_b128(unsigned lds, const void* gaddr) {
  unsigned long long ga = (unsigned long long)gaddr;
  asm volatile("global_load_async_to_lds_b128 %0, %1, off"
               :: "v"(lds), "v"(ga) : "memory");
}

// ---------------- NCHW f32 -> NHWC f16 (channel-padded with zeros) ----------
__global__ void pack_nhwc(const float* __restrict__ src, _Float16* __restrict__ dst,
                          int B, int C, int Cpad, int H, int W) {
  int idx = blockIdx.x * blockDim.x + threadIdx.x;
  int total = B * H * W * Cpad;
  if (idx >= total) return;
  int c = idx % Cpad;
  int p = idx / Cpad;
  int x = p % W; int y = (p / W) % H; int b = p / (W * H);
  float v = 0.f;
  if (c < C) v = src[(((size_t)b * C + c) * H + y) * W + x];
  dst[idx] = (_Float16)v;
}

// -------- weights OIHW f32 -> [tap][Cout][CinPad] f16 (B-matrix layout) -----
// Within every 32-channel chunk, positions hold channels in natural order,
// matching the contiguous per-lane K order (k = 16h + s) used for A, so both
// fragments are straight 32B LDS reads with identical K order.
__global__ void prep_w(const float* __restrict__ src, _Float16* __restrict__ dst,
                       int O, int I, int Ipad, int K) {
  int idx = blockIdx.x * blockDim.x + threadIdx.x;
  int total = K * K * O * Ipad;
  if (idx >= total) return;
  int ci = idx % Ipad;          // destination position
  int r  = idx / Ipad;
  int o  = r % O;
  int t  = r / O;
  int ky = t / K, kx = t % K;
  int p = ci & 31;              // position within 32-chunk
  int q = p;
  int srcci = (ci & ~31) + q;   // source channel
  float v = 0.f;
  if (srcci < I) v = src[(((size_t)o * I + srcci) * K + ky) * K + kx];
  dst[idx] = (_Float16)v;
}

// ---------------------------------------------------------------------------
// Implicit-GEMM conv, all shapes compile-time.
// Block: 128 threads (4 wave32). Tile: 32 pixels x NTILE out-channels.
// Wave wv: wm=wv&1 -> 16-pixel half; wn=wv>>1 -> channel group.
// NTILE=64: each wave owns 2 accumulators (16x32 of C); NTILE=32: 1.
// K pipeline: flattened (tap, 32-chunk) loop, double-buffered LDS fed by
// async global->LDS b128 loads; every thread issues exactly S loads/stage
// (OOB pixels read a zeroed global pad), so s_wait_asynccnt S after issuing
// stage k+1 proves stage k has landed (async loads complete in order).
// ---------------------------------------------------------------------------
template <int HIN, int CIN, int COUT, int KS, int STRIDE, int NTILE>
__global__ __launch_bounds__(128) void conv_wmma_t(
    const _Float16* __restrict__ in, const _Float16* __restrict__ wgt,
    float* __restrict__ raw, float* __restrict__ stats,
    const _Float16* __restrict__ zbuf, int do_stats)
{
  constexpr int PAD    = (KS == 3) ? 1 : 0;
  constexpr int HOUT   = (HIN + 2 * PAD - KS) / STRIDE + 1;
  constexpr int HWO    = HOUT * HOUT;
  constexpr int CHUNKS = CIN / 32;
  constexpr int KTOT   = KS * KS * CHUNKS;
  constexpr int NACC   = NTILE / 32;      // accumulators per wave
  constexpr int S      = 1 + NACC;        // async loads per thread per stage

  __shared__ alignas(32) _Float16 At[2][32][32];
  __shared__ alignas(32) _Float16 Bt[2][NTILE][32];

  const int tid  = threadIdx.x;
  const int lane = tid & 31;
  const int wv   = tid >> 5;
  const int wm   = wv & 1;
  const int wn   = wv >> 1;
  const int m0   = blockIdx.x * 32;         // pixel base (never crosses batch)
  const int n0   = blockIdx.y * NTILE;      // out-channel base

  const int pl = tid >> 2;                  // loader row 0..31
  const int qc = tid & 3;                   // 16B chunk
  const int gp = m0 + pl;
  const int bb = gp / HWO;                  // power-of-two -> shifts
  const int rp = gp - bb * HWO;
  const int oy = rp / HOUT;
  const int ox = rp - oy * HOUT;

  auto stage = [&](int k, int s) {
    const int t  = k / CHUNKS;
    const int c0 = (k - t * CHUNKS) * 32;
    int yi, xi;
    if constexpr (KS == 3) {
      const int ky = t / 3, kx = t - ky * 3;
      yi = oy * STRIDE + ky - 1;
      xi = ox * STRIDE + kx - 1;
    } else {
      yi = oy; xi = ox;
    }
    const bool inb = ((unsigned)yi < (unsigned)HIN) & ((unsigned)xi < (unsigned)HIN);
    const _Float16* ga = inb
        ? in + ((size_t)(bb * HIN + yi) * HIN + xi) * CIN + c0 + qc * 8
        : zbuf + qc * 8;
    async_b128((unsigned)(unsigned long long)&At[s][pl][qc * 8], ga);
#pragma unroll
    for (int l = 0; l < NACC; ++l) {
      const int row = pl + 32 * l;
      const _Float16* gb =
          wgt + ((size_t)t * COUT + (n0 + row)) * CIN + c0 + qc * 8;
      async_b128((unsigned)(unsigned long long)&Bt[s][row][qc * 8], gb);
    }
  };

  v8f acc[NACC] = {};
  const int fm = lane & 15;
  const int fh = lane >> 4;
  const int cbase = (NTILE == 64) ? wn * 32 : wn * 16;

  stage(0, 0);
  for (int k = 0; k < KTOT; ++k) {
    const int s = k & 1;
    if (k + 1 < KTOT) { stage(k + 1, s ^ 1); wait_async<S>(); }
    else              { wait_async<0>(); }
    __syncthreads();

    // Single contiguous 32B fragment loads (contiguous k = 16h + s order on both sides)
    v16h Af = *(const v16h*)&At[s][wm * 16 + fm][fh * 16];
#pragma unroll
    for (int j = 0; j < NACC; ++j) {
      const int row = cbase + j * 16 + fm;
      v16h Bf = *(const v16h*)&Bt[s][row][fh * 16];
      acc[j] = __builtin_amdgcn_wmma_f32_16x16x32_f16(
          false, Af, false, Bf, (short)0, acc[j], false, false);
    }
    if (k + 1 < KTOT) __syncthreads();
  }

  // ---- epilogue: raw f32 NHWC store + per-(batch,channel) stats ----
#pragma unroll
  for (int j = 0; j < NACC; ++j) {
    const int ch = n0 + cbase + j * 16 + fm;
    float sm = 0.f, sq = 0.f;
#pragma unroll
    for (int r = 0; r < 8; ++r) {
      const int pm = m0 + wm * 16 + fh * 8 + r;   // C-matrix row mapping
      const float v = acc[j][r];
      raw[(size_t)pm * COUT + ch] = v;
      sm += v; sq += v * v;
    }
    if (do_stats) {
      sm += __shfl_xor(sm, 16, 32);
      sq += __shfl_xor(sq, 16, 32);
      if (fh == 0) {
        const int bidx = m0 / HWO;
        atomicAdd(&stats[((size_t)bidx * COUT + ch) * 2 + 0], sm);
        atomicAdd(&stats[((size_t)bidx * COUT + ch) * 2 + 1], sq);
      }
    }
  }
}

// --------- instance-norm + ReLU (mode 0) or plain cast (mode 1) -> f16 ------
__global__ void norm_act(const float* __restrict__ raw, const float* __restrict__ stats,
                         const float* __restrict__ gam, const float* __restrict__ bet,
                         _Float16* __restrict__ out, int HW, int C, int total,
                         float inv_hw, int mode) {
  int idx = blockIdx.x * blockDim.x + threadIdx.x;
  if (idx >= total) return;
  int c = idx % C;
  int pix = idx / C;
  int b = pix / HW;
  float x = raw[idx];
  if (mode == 0) {
    float su = stats[((size_t)b * C + c) * 2 + 0];
    float sq = stats[((size_t)b * C + c) * 2 + 1];
    float m   = su * inv_hw;
    float var = fmaxf(sq * inv_hw - m * m, 0.f);
    float sc  = gam[c] * rsqrtf(var + 1e-5f);
    x = fmaxf((x - m) * sc + bet[c], 0.f);
  }
  out[idx] = (_Float16)x;
}

// --------- l2norm over channels; writes NCHW f32 (output) + NHWC f32 --------
__global__ void l2norm_k(const float* __restrict__ raw, float* __restrict__ out_nchw,
                         float* __restrict__ out_nhwc, int B, int H, int W, int C) {
  const int pix = blockIdx.x * blockDim.x + threadIdx.x;
  if (pix >= B * H * W) return;
  const float* p = raw + (size_t)pix * C;
  float ss = 0.f;
  for (int c = 0; c < C; ++c) { float v = p[c]; ss += v * v; }
  const float inv = 1.f / fmaxf(sqrtf(ss), 1e-6f);
  const int b = pix / (H * W);
  const int rr = pix - b * H * W;
  const int y = rr / W, x = rr - y * W;
  for (int c = 0; c < C; ++c) {
    float v = p[c] * inv;
    if (v != v) v = 0.f;                       // nan_to_num
    out_nhwc[(size_t)pix * C + c] = v;
    out_nchw[(((size_t)b * C + c) * H + y) * W + x] = v;
  }
}

// --------- translation cost: cost[b,i,j] = sum_{c,h,w} geo[clamp]*onl --------
__global__ __launch_bounds__(256) void corr_k(
    const float* __restrict__ onl, const float* __restrict__ geo,
    float* __restrict__ cost, int B, int H, int W, int C, int R)
{
  __shared__ float red[8];
  const int S = 2 * R + 1;
  const int bij = blockIdx.x;
  const int b  = bij / (S * S);
  const int rr = bij - b * S * S;
  const int di = rr / S - R;
  const int dj = rr % S - R;
  float part = 0.f;
  for (int p = threadIdx.x; p < H * W; p += blockDim.x) {
    const int h = p / W, w = p - h * W;
    const int gh = min(max(h + di, 0), H - 1);   // edge padding == clamp
    const int gw = min(max(w + dj, 0), W - 1);
    const float4* a = (const float4*)(onl + ((size_t)(b * H + h)  * W + w)  * C);
    const float4* g = (const float4*)(geo + ((size_t)(b * H + gh) * W + gw) * C);
    for (int c = 0; c < C / 4; ++c) {
      float4 x = a[c], y = g[c];
      part += x.x * y.x + x.y * y.y + x.z * y.z + x.w * y.w;
    }
  }
  const int lane = threadIdx.x & 31, wv = threadIdx.x >> 5;
  part += __shfl_down(part, 16, 32);
  part += __shfl_down(part, 8, 32);
  part += __shfl_down(part, 4, 32);
  part += __shfl_down(part, 2, 32);
  part += __shfl_down(part, 1, 32);
  if (lane == 0) red[wv] = part;
  __syncthreads();
  if (threadIdx.x == 0) {
    float sum = 0.f;
    for (int i = 0; i < 8; ++i) sum += red[i];
    cost[bij] = sum;
  }
}

// --------- 5x5 smoother (zero-pad 2) with the provided kernel ---------------
__global__ void smooth_k(const float* __restrict__ cost, const float* __restrict__ sw,
                         float* __restrict__ logits, int S, int K) {
  const int b = blockIdx.x;
  const int ij = threadIdx.x;
  if (ij >= S * S) return;
  const int i = ij / S, j = ij % S;
  const int r = K / 2;
  float acc = 0.f;
  for (int u = 0; u < K; ++u)
    for (int v = 0; v < K; ++v) {
      const int ii = i + u - r, jj = j + v - r;
      if (ii >= 0 && ii < S && jj >= 0 && jj < S)
        acc += sw[u * K + v] * cost[(b * S + ii) * S + jj];
    }
  logits[(b * S + i) * S + j] = acc;
}

// ============================= host side ====================================

struct TW {
  const float *aw, *ag, *ab;
  const float *stw, *stg, *stb;
  const float *w1[3], *g1[3], *b1[3], *w2[3], *g2[3], *b2[3];
  const float *head, *proj;
  int cin;
};

template <int HIN, int CIN, int COUT, int KS, int STRIDE, int NTILE>
static void run_conv_t(hipStream_t stream, const _Float16* actin, _Float16* actout,
                       float* RAW, float* STATS, _Float16* WGT, const _Float16* ZBUF,
                       const float* w, const float* g, const float* b,
                       int CinReal, int mode /*0 IN+ReLU, 1 cast, 2 raw only*/) {
  constexpr int B = 4;
  constexpr int PAD = (KS == 3) ? 1 : 0;
  constexpr int HO = (HIN + 2 * PAD - KS) / STRIDE + 1;
  const int wn = KS * KS * COUT * CIN;
  prep_w<<<(wn + 255) / 256, 256, 0, stream>>>(w, WGT, COUT, CinReal, CIN, KS);
  if (mode == 0)
    hipMemsetAsync(STATS, 0, (size_t)B * COUT * 2 * sizeof(float), stream);
  dim3 grid((unsigned)((B * HO * HO) / 32), (unsigned)(COUT / NTILE));
  conv_wmma_t<HIN, CIN, COUT, KS, STRIDE, NTILE>
      <<<grid, 128, 0, stream>>>(actin, WGT, RAW, STATS, ZBUF, mode == 0 ? 1 : 0);
  if (mode <= 1) {
    const int tot = B * HO * HO * COUT;
    norm_act<<<(tot + 255) / 256, 256, 0, stream>>>(RAW, STATS, g, b, actout,
                                                    HO * HO, COUT, tot,
                                                    1.0f / (float)(HO * HO), mode);
  }
}

extern "C" void kernel_launch(void* const* d_in, const int* in_sizes, int n_in,
                              void* d_out, int out_size, void* d_ws, size_t ws_size,
                              hipStream_t stream) {
  char* ws = (char*)d_ws;
  _Float16* ACT0  = (_Float16*)(ws + ((size_t)0 << 20));
  _Float16* ACT1  = (_Float16*)(ws + ((size_t)16 << 20));
  float*    RAW   = (float*)(ws + ((size_t)32 << 20));
  _Float16* WGT   = (_Float16*)(ws + ((size_t)64 << 20));
  float*    STATS = (float*)(ws + ((size_t)65 << 20));
  float*    PROJ[2] = { (float*)(ws + ((size_t)66 << 20)),
                        (float*)(ws + ((size_t)70 << 20)) };
  float*    COST  = (float*)(ws + ((size_t)74 << 20));
  _Float16* ZBUF  = (_Float16*)(ws + ((size_t)75 << 20));
  float* out = (float*)d_out;

  hipMemsetAsync(ZBUF, 0, 256, stream);   // zero pad for OOB async loads

  auto IN = [&](int i) { return (const float*)d_in[i]; };
  const float *lidar, *geo, *sw;
  TW tw[2];

  if (n_in > 0 && in_sizes[0] == 786432) {
    // jax pytree flatten (sorted dict keys): geospatial first
    geo = IN(0); lidar = IN(1);
    int k = 2;
    for (int t = 0; t < 2; ++t) {
      TW& T = tw[t];
      T.ab = IN(k++); T.ag = IN(k++); T.aw = IN(k++);
      for (int bl = 0; bl < 3; ++bl) {
        T.b1[bl] = IN(k++); T.b2[bl] = IN(k++);
        T.g1[bl] = IN(k++); T.g2[bl] = IN(k++);
        T.w1[bl] = IN(k++); T.w2[bl] = IN(k++);
      }
      T.head = IN(k++);
      T.stb = IN(k++); T.stg = IN(k++); T.stw = IN(k++);
      T.proj = IN(k++);
    }
    sw = IN(k++);
  } else {
    // insertion-order flatten: lidar first
    lidar = IN(0); geo = IN(1);
    tw[0].aw = IN(2); tw[0].ag = IN(3); tw[0].ab = IN(4);
    tw[1].aw = IN(5); tw[1].ag = IN(6); tw[1].ab = IN(7);
    int k = 8;
    for (int t = 0; t < 2; ++t) {
      TW& T = tw[t];
      T.stw = IN(k++); T.stg = IN(k++); T.stb = IN(k++);
      for (int bl = 0; bl < 3; ++bl) {
        T.w1[bl] = IN(k++); T.g1[bl] = IN(k++); T.b1[bl] = IN(k++);
        T.w2[bl] = IN(k++); T.g2[bl] = IN(k++); T.b2[bl] = IN(k++);
      }
      T.head = IN(k++);
    }
    tw[0].proj = IN(k++); tw[1].proj = IN(k++); sw = IN(k++);
  }
  tw[0].cin = 4;  // lidar channels
  tw[1].cin = 3;  // geospatial channels
  const float* xin[2] = { lidar, geo };

  const int B = 4;
  const size_t projN = (size_t)B * 64 * 64 * 64;   // 1,048,576

  for (int t = 0; t < 2; ++t) {
    const TW& T = tw[t];
    // pack input NCHW -> NHWC f16 padded to 32 channels
    {
      const int total = B * 256 * 256 * 32;
      pack_nhwc<<<(total + 255) / 256, 256, 0, stream>>>(xin[t], ACT0, B, T.cin, 32, 256, 256);
    }
    // adapter 3x3 (cin->32), IN+ReLU
    run_conv_t<256, 32, 32, 3, 1, 32>(stream, ACT0, ACT1, RAW, STATS, WGT, ZBUF,
                                      T.aw, T.ag, T.ab, T.cin, 0);
    // stem 3x3 32->32
    run_conv_t<256, 32, 32, 3, 1, 32>(stream, ACT1, ACT0, RAW, STATS, WGT, ZBUF,
                                      T.stw, T.stg, T.stb, 32, 0);
    // block0: 32->64 s2 (256->128), then 64->64
    run_conv_t<256, 32, 64, 3, 2, 64>(stream, ACT0, ACT1, RAW, STATS, WGT, ZBUF,
                                      T.w1[0], T.g1[0], T.b1[0], 32, 0);
    run_conv_t<128, 64, 64, 3, 1, 64>(stream, ACT1, ACT0, RAW, STATS, WGT, ZBUF,
                                      T.w2[0], T.g2[0], T.b2[0], 64, 0);
    // block1: 64->128 s2 (128->64), then 128->128
    run_conv_t<128, 64, 128, 3, 2, 64>(stream, ACT0, ACT1, RAW, STATS, WGT, ZBUF,
                                       T.w1[1], T.g1[1], T.b1[1], 64, 0);
    run_conv_t<64, 128, 128, 3, 1, 64>(stream, ACT1, ACT0, RAW, STATS, WGT, ZBUF,
                                       T.w2[1], T.g2[1], T.b2[1], 128, 0);
    // block2: 128->128 s1, twice
    run_conv_t<64, 128, 128, 3, 1, 64>(stream, ACT0, ACT1, RAW, STATS, WGT, ZBUF,
                                       T.w1[2], T.g1[2], T.b1[2], 128, 0);
    run_conv_t<64, 128, 128, 3, 1, 64>(stream, ACT1, ACT0, RAW, STATS, WGT, ZBUF,
                                       T.w2[2], T.g2[2], T.b2[2], 128, 0);
    // head 1x1 128->128, no norm (plain cast to f16)
    run_conv_t<64, 128, 128, 1, 1, 64>(stream, ACT0, ACT1, RAW, STATS, WGT, ZBUF,
                                       T.head, nullptr, nullptr, 128, 1);
    // proj 1x1 128->64, raw f32 only
    run_conv_t<64, 128, 64, 1, 1, 64>(stream, ACT1, nullptr, RAW, STATS, WGT, ZBUF,
                                      T.proj, nullptr, nullptr, 128, 2);
    // l2norm: RAW -> d_out NCHW region + PROJ[t] NHWC
    {
      float* dst_nchw = out + 1156 + (size_t)t * projN;
      const int pixels = B * 64 * 64;
      l2norm_k<<<(pixels + 255) / 256, 256, 0, stream>>>(RAW, dst_nchw, PROJ[t], B, 64, 64, 64);
    }
  }

  // translation cost over 17x17 shifts, then 5x5 smoother -> logits
  corr_k<<<B * 17 * 17, 256, 0, stream>>>(PROJ[0], PROJ[1], COST, B, 64, 64, 64, 8);
  smooth_k<<<B, 17 * 17, 0, stream>>>(COST, sw, out, 17, 5);

  (void)n_in; (void)out_size; (void)ws_size;
}
